// GNNInLoopSlotAttention_57294863728796
// MI455X (gfx1250) — matrix operand order repaired and checked by reading, not compiled, not run on hardware
//
#include <hip/hip_runtime.h>
#include <hip/hip_bf16.h>

typedef __bf16 bf16;
typedef __attribute__((ext_vector_type(16))) bf16 v16bf;
typedef __attribute__((ext_vector_type(8)))  bf16 v8bf;
typedef __attribute__((ext_vector_type(8)))  float v8f;

#define DEV static __device__ __forceinline__

constexpr int cB = 32, cN = 4096, cD = 256, cK = 8;
constexpr int cBN = cB * cN;          // 131072 rows
constexpr size_t KV_ELEMS = (size_t)cBN * cD;

DEV v8f vzero() {
  v8f z;
#pragma unroll
  for (int i = 0; i < 8; ++i) z[i] = 0.0f;
  return z;
}

DEV v8f wmma_bf16(v16bf a, v16bf b, v8f c) {
  // D = A(16x32 bf16) * B(32x16 bf16) + C(16x16 f32)
  return __builtin_amdgcn_wmma_f32_16x16x32_bf16(false, a, false, b, (short)0, c, false, false);
}

// A fragment (16x32 bf16) from row-major bf16 [16 x stride] matrix.
// lane<16: M=lane, K in kk+0..15; lane>=16: M=lane-16, K in kk+16..31.
DEV v16bf load_a(const bf16* __restrict__ base, int stride, int kk, int lane) {
  int m  = lane & 15;
  int cb = (lane & 16) ? 16 : 0;
  const bf16* p = base + (size_t)m * stride + kk + cb;
  v8bf lo = *(const v8bf*)(p);
  v8bf hi = *(const v8bf*)(p + 8);
  v16bf a;
#pragma unroll
  for (int i = 0; i < 8; ++i) { a[i] = lo[i]; a[i + 8] = hi[i]; }
  return a;
}

// ---------------------------------------------------------------- pack 256x256 f32 weight into WMMA-B fragment-linear bf16
// layout: P[((nt*8+kt)*32+lane)*16 + j] = W[kt*32 + cb + j][nt*16 + (lane&15)], cb = lane<16 ? 0 : 16
__global__ __launch_bounds__(256) void k_pack(const float* __restrict__ W, bf16* __restrict__ P) {
  int tid = blockIdx.x * 256 + threadIdx.x;      // 65536
  int j = tid & 15, lane = (tid >> 4) & 31, kt = (tid >> 9) & 7, nt = tid >> 12;
  int cb = (lane & 16) ? 16 : 0;
  P[tid] = (bf16)W[(kt * 32 + cb + j) * 256 + nt * 16 + (lane & 15)];
}

// ---------------------------------------------------------------- init: slots = mu + sigma*noise, zero pad rows
__global__ __launch_bounds__(256) void k_init(const float* __restrict__ mu,
                                              const float* __restrict__ sigma,
                                              const float* __restrict__ noise,
                                              float* __restrict__ slots,
                                              bf16* __restrict__ attn_bf,
                                              bf16* __restrict__ qbf) {
  int tid = blockIdx.x * 256 + threadIdx.x;      // 1,048,576 threads
  {
    int b = tid >> 15, r = (tid >> 12) & 7, j = tid & 4095;
    attn_bf[(size_t)b * 65536 + (size_t)(8 + r) * 4096 + j] = (bf16)0.f;
  }
  if (tid < 65536) {
    int b = tid >> 11, r = (tid >> 8) & 7, d = tid & 255;
    qbf[(size_t)b * 4096 + (size_t)(8 + r) * 256 + d] = (bf16)0.f;
    slots[tid] = mu[tid & 2047] + sigma[tid & 2047] * noise[tid];
  }
}

// ---------------------------------------------------------------- fused LN-stats + LN + K/V projection (WMMA)
// Tile of 16 rows staged into LDS via async-to-LDS loads; per-row stats computed in LDS;
// 8 waves: waves 0-3 -> k cols [w*64,...), waves 4-7 -> v cols.
__global__ __launch_bounds__(256) void k_proj_kv(
    const float* __restrict__ x, const float* __restrict__ lns,
    const float* __restrict__ lnb, const bf16* __restrict__ packK,
    const bf16* __restrict__ packV, const float* __restrict__ kbias,
    const float* __restrict__ vbias, bf16* __restrict__ kbf, bf16* __restrict__ vt) {
  __shared__ float xt[16 * 256];          // 16KB input tile
  __shared__ float red1[256], red2[256];
  __shared__ float smean[16], srstd[16];
  __shared__ float sLn[256], sLb[256];

  int t = threadIdx.x;
  int lane = t & 31;
  int wave = t >> 5;
  int rowbase = blockIdx.x * 16;

  // ---- stage 16x256 f32 tile into LDS with gfx1250 async-to-LDS copies ----
  {
    const float* gsrc = x + (size_t)rowbase * cD;
    unsigned ldsbase = (unsigned)(uintptr_t)(&xt[0]);
#pragma unroll
    for (int c = 0; c < 4; ++c) {
      unsigned lofs = ldsbase + (unsigned)(t * 64 + c * 16);   // bytes
      const float* ga = gsrc + t * 16 + c * 4;
      asm volatile("global_load_async_to_lds_b128 %0, %1, off"
                   :: "v"(lofs), "v"(ga) : "memory");
    }
    sLn[t] = lns[t];
    sLb[t] = lnb[t];
    asm volatile("s_wait_asynccnt 0x0" ::: "memory");
  }
  __syncthreads();

  // ---- per-row mean / rstd (16 rows, 16 partials each) ----
  {
    int row16 = t >> 4, seg = t & 15;
    float p1 = 0.f, p2 = 0.f;
#pragma unroll
    for (int u = 0; u < 16; ++u) {
      float vv = xt[row16 * 256 + seg * 16 + u];
      p1 += vv; p2 += vv * vv;
    }
    red1[t] = p1; red2[t] = p2;
  }
  __syncthreads();
  if (t < 16) {
    float s1 = 0.f, s2 = 0.f;
#pragma unroll
    for (int u = 0; u < 16; ++u) { s1 += red1[t * 16 + u]; s2 += red2[t * 16 + u]; }
    float m = s1 * (1.f / 256.f);
    smean[t] = m;
    srstd[t] = rsqrtf(s2 * (1.f / 256.f) - m * m + 1e-5f);
  }
  __syncthreads();

  // ---- WMMA GEMM against packed weights ----
  bool isV = wave >= 4;
  int colbase = (wave & 3) * 64;
  const bf16* pack = isV ? packV : packK;
  const float* bias = isV ? vbias : kbias;

  int m  = lane & 15;
  int cb = (lane & 16) ? 16 : 0;
  float mu = smean[m], rs = srstd[m];
  const float* xrow = &xt[m * 256];

  v8f acc[4];
#pragma unroll
  for (int c = 0; c < 4; ++c) acc[c] = vzero();

  for (int kt = 0; kt < 8; ++kt) {
    int kk = kt * 32;
    v16bf a;
#pragma unroll
    for (int half = 0; half < 2; ++half) {
      int k0 = kk + cb + half * 8;
      float4 f0 = *(const float4*)(xrow + k0);
      float4 f1 = *(const float4*)(xrow + k0 + 4);
      float xv[8] = {f0.x, f0.y, f0.z, f0.w, f1.x, f1.y, f1.z, f1.w};
#pragma unroll
      for (int i = 0; i < 8; ++i) {
        float ln = (xv[i] - mu) * rs * sLn[k0 + i] + sLb[k0 + i];
        a[half * 8 + i] = (bf16)ln;
      }
    }
#pragma unroll
    for (int c = 0; c < 4; ++c) {
      int nt = (colbase >> 4) + c;
      v16bf bfrag = *(const v16bf*)(pack + ((size_t)(nt * 8 + kt) * 32 + lane) * 16);
      acc[c] = wmma_bf16(a, bfrag, acc[c]);
    }
  }
#pragma unroll
  for (int c = 0; c < 4; ++c) {
#pragma unroll
    for (int r = 0; r < 8; ++r) {
      int token = rowbase + r + ((lane & 16) ? 8 : 0);
      int d = colbase + c * 16 + (lane & 15);
      float val = acc[c][r] + bias[d];
      if (!isV) {
        kbf[(size_t)token * 256 + d] = (bf16)val;                     // (BN, D) row-major
      } else {
        int b = token >> 12, tj = token & 4095;
        vt[(size_t)b * 1048576 + (size_t)d * 4096 + tj] = (bf16)val;  // (B, D, N) transposed
      }
    }
  }
}

// ---------------------------------------------------------------- q = LN(slots) @ qw + qb  (tiny, VALU)
__global__ __launch_bounds__(256) void k_qproj(const float* __restrict__ slots,
                                               const float* __restrict__ lnS,
                                               const float* __restrict__ lnB,
                                               const float* __restrict__ qw,
                                               const float* __restrict__ qb,
                                               bf16* __restrict__ qbf) {
  int b = blockIdx.x >> 3, i = blockIdx.x & 7, t = threadIdx.x;
  __shared__ float s1[256], s2[256], arr[256];
  float v = slots[((size_t)b * 8 + i) * 256 + t];
  s1[t] = v; s2[t] = v * v;
  __syncthreads();
  for (int s = 128; s > 0; s >>= 1) {
    if (t < s) { s1[t] += s1[t + s]; s2[t] += s2[t + s]; }
    __syncthreads();
  }
  float m = s1[0] * (1.f / 256.f);
  float rs = rsqrtf(s2[0] * (1.f / 256.f) - m * m + 1e-5f);
  arr[t] = (v - m) * rs * lnS[t] + lnB[t];
  __syncthreads();
  float acc = qb[t];
  for (int e = 0; e < 256; ++e) acc += arr[e] * qw[e * 256 + t];
  qbf[(size_t)b * 4096 + (size_t)i * 256 + t] = (bf16)acc;
}

// ---------------------------------------------------------------- dots = q @ k^T (WMMA) fused with softmax over slots + EPS
__global__ __launch_bounds__(256) void k_dots(const bf16* __restrict__ qbf,
                                              const bf16* __restrict__ kbf,
                                              float* __restrict__ attn_un) {
  int lane = threadIdx.x & 31;
  int wave = threadIdx.x >> 5;
  int job = blockIdx.x * 8 + wave;   // 8192 jobs
  int b = job >> 8;
  int jt = job & 255;
  const bf16* abase = qbf + (size_t)b * 4096;
  int cbB = (lane & 16) ? 16 : 0;
  const bf16* brow = kbf + ((size_t)b * 4096 + jt * 16 + (lane & 15)) * 256 + cbB;
  v8f acc = vzero();
  for (int kt = 0; kt < 8; ++kt) {
    v16bf a = load_a(abase, 256, kt * 32, lane);
    v16bf bfrag = *(const v16bf*)(brow + kt * 32);
    acc = wmma_bf16(a, bfrag, acc);
  }
  // lanes 0-15 hold all 8 slot scores for token jt*16+lane in acc[0..7]
  float vals[8];
#pragma unroll
  for (int i = 0; i < 8; ++i) vals[i] = acc[i] * 0.0625f;   // * D^-0.5
  float mx = vals[0];
#pragma unroll
  for (int i = 1; i < 8; ++i) mx = fmaxf(mx, vals[i]);
  float sum = 0.f;
#pragma unroll
  for (int i = 0; i < 8; ++i) { vals[i] = __expf(vals[i] - mx); sum += vals[i]; }
  float inv = 1.f / sum;
  if (lane < 16) {
    int j = jt * 16 + lane;
#pragma unroll
    for (int i = 0; i < 8; ++i)
      attn_un[((size_t)b * 8 + i) * 4096 + j] = vals[i] * inv + 1e-8f;
  }
}

// ---------------------------------------------------------------- deterministic row sums over tokens
__global__ __launch_bounds__(256) void k_rowsum(const float* __restrict__ attn_un,
                                                float* __restrict__ rowsum) {
  int row = blockIdx.x, t = threadIdx.x;
  __shared__ float r[256];
  float s = 0.f;
  const float* ar = attn_un + (size_t)row * 4096;
  for (int j = t; j < 4096; j += 256) s += ar[j];
  r[t] = s;
  __syncthreads();
  for (int ss = 128; ss > 0; ss >>= 1) {
    if (t < ss) r[t] += r[t + ss];
    __syncthreads();
  }
  if (t == 0) rowsum[row] = r[0];
}

// ---------------------------------------------------------------- attn normalize -> f32 out + bf16 A-matrix
__global__ __launch_bounds__(256) void k_norm(const float* __restrict__ attn_un,
                                              const float* __restrict__ rowsum,
                                              float* __restrict__ attn_out,
                                              bf16* __restrict__ attn_bf) {
  int idx = blockIdx.x * 256 + threadIdx.x;   // 32*8*4096
  int b = idx >> 15;
  int r = idx & 32767;
  int i = r >> 12;
  int j = r & 4095;
  float a = attn_un[idx] / rowsum[b * 8 + i];
  attn_out[idx] = a;
  attn_bf[(size_t)b * 65536 + (size_t)i * 4096 + j] = (bf16)a;
}

// ---------------------------------------------------------------- updates = attn_n @ v (WMMA, split-K x4 for bandwidth)
__global__ __launch_bounds__(256) void k_updates(const bf16* __restrict__ attn_bf,
                                                 const bf16* __restrict__ vt,
                                                 float* __restrict__ up_part) {
  int lane = threadIdx.x & 31;
  int wave = threadIdx.x >> 5;
  int job = blockIdx.x * 8 + wave;  // 2048 jobs
  int chunk = job & 3;
  int dt = (job >> 2) & 15;
  int b = job >> 6;
  const bf16* abase = attn_bf + (size_t)b * 65536;
  int cbB = (lane & 16) ? 16 : 0;
  const bf16* brow = vt + (size_t)b * 1048576 + (size_t)(dt * 16 + (lane & 15)) * 4096 + cbB;
  v8f acc = vzero();
  for (int kt = chunk * 32; kt < chunk * 32 + 32; ++kt) {
    v16bf a = load_a(abase, 4096, kt * 32, lane);
    v16bf bfrag = *(const v16bf*)(brow + kt * 32);
    acc = wmma_bf16(a, bfrag, acc);
  }
  if (lane < 16) {
    int d = dt * 16 + lane;
#pragma unroll
    for (int r = 0; r < 8; ++r)
      up_part[((size_t)chunk * 256 + (size_t)b * 8 + r) * 256 + d] = acc[r];
  }
}

// ---------------------------------------------------------------- GRU update, in-place on slots (256 rows)
__global__ __launch_bounds__(256) void k_gru(const float* __restrict__ up_part,
                                             float* __restrict__ slots,
                                             const float* __restrict__ wih,
                                             const float* __restrict__ whh,
                                             const float* __restrict__ bih,
                                             const float* __restrict__ bhh) {
  int row = blockIdx.x, t = threadIdx.x;
  __shared__ float xs[256], hs[256];
  xs[t] = up_part[((size_t)0 * 256 + row) * 256 + t]
        + up_part[((size_t)1 * 256 + row) * 256 + t]
        + up_part[((size_t)2 * 256 + row) * 256 + t]
        + up_part[((size_t)3 * 256 + row) * 256 + t];
  float h = slots[(size_t)row * 256 + t];
  hs[t] = h;
  __syncthreads();
  float gi[3], gh[3];
#pragma unroll
  for (int g = 0; g < 3; ++g) {
    int o = g * 256 + t;
    float ai = bih[o], ah = bhh[o];
    const float* wi = wih + (size_t)o * 256;
    const float* wh = whh + (size_t)o * 256;
    for (int e = 0; e < 256; ++e) { ai += xs[e] * wi[e]; ah += hs[e] * wh[e]; }
    gi[g] = ai; gh[g] = ah;
  }
  float r = 1.f / (1.f + __expf(-(gi[0] + gh[0])));
  float z = 1.f / (1.f + __expf(-(gi[1] + gh[1])));
  float n = tanhf(gi[2] + r * gh[2]);
  slots[(size_t)row * 256 + t] = (1.f - z) * n + z * h;
}

// ---------------------------------------------------------------- residual MLP (256 rows)
__global__ __launch_bounds__(256) void k_mlp(float* __restrict__ slots,
                                             const float* __restrict__ lnS,
                                             const float* __restrict__ lnB,
                                             const float* __restrict__ w1,
                                             const float* __restrict__ b1,
                                             const float* __restrict__ w2,
                                             const float* __restrict__ b2) {
  int row = blockIdx.x, t = threadIdx.x;
  __shared__ float s1[256], s2[256], hln[256], hid[1024];
  float v = slots[(size_t)row * 256 + t];
  s1[t] = v; s2[t] = v * v;
  __syncthreads();
  for (int s = 128; s > 0; s >>= 1) {
    if (t < s) { s1[t] += s1[t + s]; s2[t] += s2[t + s]; }
    __syncthreads();
  }
  float m = s1[0] * (1.f / 256.f);
  float rs = rsqrtf(s2[0] * (1.f / 256.f) - m * m + 1e-5f);
  hln[t] = (v - m) * rs * lnS[t] + lnB[t];
  __syncthreads();
#pragma unroll
  for (int u = 0; u < 4; ++u) {
    int o = u * 256 + t;
    float a = b1[o];
    for (int e = 0; e < 256; ++e) a += hln[e] * w1[(size_t)e * 1024 + o];
    hid[o] = fmaxf(a, 0.f);
  }
  __syncthreads();
  float a = b2[t];
  for (int hh = 0; hh < 1024; ++hh) a += hid[hh] * w2[(size_t)hh * 256 + t];
  slots[(size_t)row * 256 + t] = v + a;
}

// ---------------------------------------------------------------- positions = a_n @ coords (reduction per (b,slot))
__global__ __launch_bounds__(256) void k_positions(const float* __restrict__ attn,
                                                   float* __restrict__ positions) {
  int row = blockIdx.x, t = threadIdx.x;
  __shared__ float rs[256], ry[256], rx[256];
  float s = 0, sy = 0, sx = 0;
  const float* ar = attn + (size_t)row * 4096;
  for (int j = t; j < 4096; j += 256) {
    float a = ar[j];
    float yy = (float)(j >> 6) * (1.f / 63.f);
    float xx = (float)(j & 63) * (1.f / 63.f);
    s += a; sy += a * yy; sx += a * xx;
  }
  rs[t] = s; ry[t] = sy; rx[t] = sx;
  __syncthreads();
  for (int ss = 128; ss > 0; ss >>= 1) {
    if (t < ss) { rs[t] += rs[t + ss]; ry[t] += ry[t + ss]; rx[t] += rx[t + ss]; }
    __syncthreads();
  }
  if (t == 0) {
    float inv = 1.f / (rs[0] + 1e-8f);
    positions[row * 2 + 0] = ry[0] * inv;   // y (coords channel 0)
    positions[row * 2 + 1] = rx[0] * inv;   // x (coords channel 1)
  }
}

// ---------------------------------------------------------------- GNN over 8 slots, one block per batch
__global__ __launch_bounds__(256) void k_gnn(
    float* __restrict__ slots, const float* __restrict__ positions,
    const float* __restrict__ gqw, const float* __restrict__ gqb,
    const float* __restrict__ gkw, const float* __restrict__ gkb,
    const float* __restrict__ gvw, const float* __restrict__ gvb,
    const float* __restrict__ e1w, const float* __restrict__ e1b,
    const float* __restrict__ e2w, const float* __restrict__ e2b,
    const float* __restrict__ gow, const float* __restrict__ gob,
    const float* __restrict__ glns, const float* __restrict__ glnb) {
  int b = blockIdx.x, t = threadIdx.x;
  __shared__ float S[8 * 256], Q[8 * 256], Km[8 * 256], V[8 * 256], O[8 * 256], T[8 * 256];
  __shared__ float norms[8], biasb[8 * 8 * 4], sc[256];
  __shared__ float red1[256], red2[256], mrow[8], rrow[8];

  const float* sb = slots + (size_t)b * 2048;
#pragma unroll
  for (int i = 0; i < 8; ++i) S[i * 256 + t] = sb[i * 256 + t];
  __syncthreads();
  if (t < 8) {
    float acc = 0.f;
    for (int e = 0; e < 256; ++e) { float vv = S[t * 256 + e]; acc += vv * vv; }
    norms[t] = sqrtf(acc);
  }
#pragma unroll 1
  for (int i = 0; i < 8; ++i) {
    float aq = gqb[t], ak = gkb[t], av = gvb[t];
    for (int e = 0; e < 256; ++e) {
      float vv = S[i * 256 + e];
      aq += vv * gqw[e * 256 + t];
      ak += vv * gkw[e * 256 + t];
      av += vv * gvw[e * 256 + t];
    }
    Q[i * 256 + t] = aq; Km[i * 256 + t] = ak; V[i * 256 + t] = av;
  }
  __syncthreads();
  if (t < 64) {  // edge features + bias MLP for pair (i,j)
    int i = t >> 3, j = t & 7;
    float pyi = positions[(b * 8 + i) * 2], pxi = positions[(b * 8 + i) * 2 + 1];
    float pyj = positions[(b * 8 + j) * 2], pxj = positions[(b * 8 + j) * 2 + 1];
    float dy = pyj - pyi, dx = pxj - pxi;
    float dsq = dy * dy + dx * dx;
    float dist = sqrtf(dsq + 1e-12f);
    float angle = (dsq > 0.f) ? atan2f(dx, dy) : 0.f;
    float dot = 0.f;
    for (int e = 0; e < 256; ++e) dot += S[i * 256 + e] * S[j * 256 + e];
    float sim = dot / (fmaxf(norms[i], 1e-12f) * fmaxf(norms[j], 1e-12f));
    float f[8] = {dy, dx, dist, angle, sim,
                  dist < 0.3f ? 1.f : 0.f, dist < 0.15f ? 1.f : 0.f, dist < 0.5f ? 1.f : 0.f};
    float bh[4] = {e2b[0], e2b[1], e2b[2], e2b[3]};
    for (int u = 0; u < 32; ++u) {
      float hv = e1b[u];
#pragma unroll
      for (int ff = 0; ff < 8; ++ff) hv += f[ff] * e1w[ff * 32 + u];
      hv = fmaxf(hv, 0.f);
#pragma unroll
      for (int hh = 0; hh < 4; ++hh) bh[hh] += hv * e2w[u * 4 + hh];
    }
#pragma unroll
    for (int hh = 0; hh < 4; ++hh) biasb[(i * 8 + j) * 4 + hh] = bh[hh];
  }
  __syncthreads();
  {  // scores: t -> (h, qi, kj)
    int h = t >> 6, qi = (t >> 3) & 7, kj = t & 7;
    float a = 0.f;
    for (int dd = 0; dd < 64; ++dd) a += Q[qi * 256 + h * 64 + dd] * Km[kj * 256 + h * 64 + dd];
    sc[t] = a * 0.125f + biasb[(qi * 8 + kj) * 4 + h];   // HD^-0.5 = 1/8
  }
  __syncthreads();
  if (t < 32) {  // softmax over kj per (h, qi)
    int base = ((t >> 3) << 6) | ((t & 7) << 3);
    float mx = sc[base];
    for (int kj = 1; kj < 8; ++kj) mx = fmaxf(mx, sc[base + kj]);
    float e[8]; float sum = 0.f;
    for (int kj = 0; kj < 8; ++kj) { e[kj] = __expf(sc[base + kj] - mx); sum += e[kj]; }
    float inv = 1.f / sum;
    for (int kj = 0; kj < 8; ++kj) sc[base + kj] = e[kj] * inv;
  }
  __syncthreads();
  {  // out = attn @ V
    int h = t >> 6;
#pragma unroll 1
    for (int qi = 0; qi < 8; ++qi) {
      float a = 0.f;
#pragma unroll
      for (int kj = 0; kj < 8; ++kj) a += sc[(h << 6) | (qi << 3) | kj] * V[kj * 256 + t];
      O[qi * 256 + t] = a;
    }
  }
  __syncthreads();
#pragma unroll 1
  for (int qi = 0; qi < 8; ++qi) {  // out proj + residual
    float a = gob[t];
    for (int e = 0; e < 256; ++e) a += O[qi * 256 + e] * gow[e * 256 + t];
    T[qi * 256 + t] = S[qi * 256 + t] + a;
  }
  __syncthreads();
  {  // LN per slot row
    int rowq = t >> 5, kq = t & 31;
    float s = 0.f, s2v = 0.f;
    for (int u = 0; u < 8; ++u) { float vv = T[rowq * 256 + kq * 8 + u]; s += vv; s2v += vv * vv; }
    red1[t] = s; red2[t] = s2v;
  }
  __syncthreads();
  if (t < 8) {
    float s = 0.f, s2v = 0.f;
    for (int u = 0; u < 32; ++u) { s += red1[t * 32 + u]; s2v += red2[t * 32 + u]; }
    float m = s * (1.f / 256.f);
    mrow[t] = m;
    rrow[t] = rsqrtf(s2v * (1.f / 256.f) - m * m + 1e-5f);
  }
  __syncthreads();
#pragma unroll
  for (int qi = 0; qi < 8; ++qi) {
    float vv = (T[qi * 256 + t] - mrow[qi]) * rrow[qi] * glns[t] + glnb[t];
    slots[(size_t)b * 2048 + qi * 256 + t] = vv;
  }
}

// ================================================================ host
extern "C" void kernel_launch(void* const* d_in, const int* in_sizes, int n_in,
                              void* d_out, int out_size, void* d_ws, size_t ws_size,
                              hipStream_t stream) {
  const float* inputs  = (const float*)d_in[0];
  const float* noise   = (const float*)d_in[1];
  const float* mu      = (const float*)d_in[2];
  const float* sigma   = (const float*)d_in[3];
  const float* qw      = (const float*)d_in[4];
  const float* qb      = (const float*)d_in[5];
  const float* kw      = (const float*)d_in[6];
  const float* kb      = (const float*)d_in[7];
  const float* vw      = (const float*)d_in[8];
  const float* vb      = (const float*)d_in[9];
  const float* gru_wih = (const float*)d_in[10];
  const float* gru_whh = (const float*)d_in[11];
  const float* gru_bih = (const float*)d_in[12];
  const float* gru_bhh = (const float*)d_in[13];
  const float* mlp_w1  = (const float*)d_in[14];
  const float* mlp_b1  = (const float*)d_in[15];
  const float* mlp_w2  = (const float*)d_in[16];
  const float* mlp_b2  = (const float*)d_in[17];
  const float* ln_in_s = (const float*)d_in[18];
  const float* ln_in_b = (const float*)d_in[19];
  const float* ln_s_s  = (const float*)d_in[20];
  const float* ln_s_b  = (const float*)d_in[21];
  const float* ln_ff_s = (const float*)d_in[22];
  const float* ln_ff_b = (const float*)d_in[23];
  const float* g_qw    = (const float*)d_in[24];
  const float* g_qb    = (const float*)d_in[25];
  const float* g_kw    = (const float*)d_in[26];
  const float* g_kb    = (const float*)d_in[27];
  const float* g_vw    = (const float*)d_in[28];
  const float* g_vb    = (const float*)d_in[29];
  const float* e1w     = (const float*)d_in[30];
  const float* e1b     = (const float*)d_in[31];
  const float* e2w     = (const float*)d_in[32];
  const float* e2b     = (const float*)d_in[33];
  const float* g_ow    = (const float*)d_in[34];
  const float* g_ob    = (const float*)d_in[35];
  const float* g_ln_s  = (const float*)d_in[36];
  const float* g_ln_b  = (const float*)d_in[37];

  char* p = (char*)d_ws;
  auto alloc = [&](size_t bytes) -> void* {
    void* r = (void*)p;
    p += (bytes + 255) & ~(size_t)255;
    return r;
  };
  bf16*  packK    = (bf16*)alloc(65536 * 2);
  bf16*  packV    = (bf16*)alloc(65536 * 2);
  bf16*  kbf      = (bf16*)alloc(KV_ELEMS * 2);
  bf16*  vt       = (bf16*)alloc(KV_ELEMS * 2);
  bf16*  qbf      = (bf16*)alloc((size_t)cB * 16 * 256 * 2);
  float* attn_un  = (float*)alloc((size_t)cB * 8 * 4096 * 4);
  float* rowsum   = (float*)alloc(256 * 4);
  bf16*  attn_bf  = (bf16*)alloc((size_t)cB * 16 * 4096 * 2);
  float* up_part  = (float*)alloc((size_t)4 * 256 * 256 * 4);
  float* slots    = (float*)alloc(65536 * 4);
  float* positions= (float*)alloc(512 * 4);

  float* out_slots = (float*)d_out;
  float* out_attn  = out_slots + 65536;

  k_pack<<<256, 256, 0, stream>>>(kw, packK);
  k_pack<<<256, 256, 0, stream>>>(vw, packV);
  k_init<<<4096, 256, 0, stream>>>(mu, sigma, noise, slots, attn_bf, qbf);
  k_proj_kv<<<cBN / 16, 256, 0, stream>>>(inputs, ln_in_s, ln_in_b,
                                          packK, packV, kb, vb, kbf, vt);
  for (int it = 0; it < 3; ++it) {
    k_qproj<<<256, 256, 0, stream>>>(slots, ln_s_s, ln_s_b, qw, qb, qbf);
    k_dots<<<1024, 256, 0, stream>>>(qbf, kbf, attn_un);
    k_rowsum<<<256, 256, 0, stream>>>(attn_un, rowsum);
    k_norm<<<4096, 256, 0, stream>>>(attn_un, rowsum, out_attn, attn_bf);
    k_updates<<<256, 256, 0, stream>>>(attn_bf, vt, up_part);
    k_gru<<<256, 256, 0, stream>>>(up_part, slots, gru_wih, gru_whh, gru_bih, gru_bhh);
    k_mlp<<<256, 256, 0, stream>>>(slots, ln_ff_s, ln_ff_b, mlp_w1, mlp_b1, mlp_w2, mlp_b2);
    if (it >= 1) {
      k_positions<<<256, 256, 0, stream>>>(out_attn, positions);
      k_gnn<<<32, 256, 0, stream>>>(slots, positions, g_qw, g_qb, g_kw, g_kb,
                                    g_vw, g_vb, e1w, e1b, e2w, e2b,
                                    g_ow, g_ob, g_ln_s, g_ln_b);
    }
  }
  hipMemcpyAsync(d_out, slots, 65536 * 4, hipMemcpyDeviceToDevice, stream);
}
